// PerformerAttention_78245714199331
// MI455X (gfx1250) — matrix operand order repaired and checked by reading, not compiled, not run on hardware
//
#include <hip/hip_runtime.h>

typedef _Float16 h16;
typedef __attribute__((ext_vector_type(16))) _Float16 v16h;
typedef __attribute__((ext_vector_type(8)))  _Float16 v8h;
typedef __attribute__((ext_vector_type(8)))  float    v8f;
typedef int v4i __attribute__((__vector_size__(4 * sizeof(int))));

#define WMMA_F32_F16(A,B,C) __builtin_amdgcn_wmma_f32_16x16x32_f16(false,(A),false,(B),(short)0,(C),false,false)

static constexpr int PB  = 2;     // batch
static constexpr int PH  = 16;    // heads
static constexpr int PT  = 1024;  // seq len
static constexpr int PD  = 64;    // head dim == M
static constexpr int PC  = 1024;  // d_model
static constexpr int PNC = 16;    // chunks of 64

#if __has_builtin(__builtin_amdgcn_global_load_async_to_lds_b128) && \
    __has_builtin(__builtin_amdgcn_s_wait_asynccnt)
#define PERF_ASYNC_LDS 1
#else
#define PERF_ASYNC_LDS 0
#endif

#define AS1 __attribute__((address_space(1)))
#define AS3 __attribute__((address_space(3)))

union AFrag { v16h v; v8h h2[2]; };

// A-fragment loader (16-bit A 16x32 layout): lane holds row (lane&15),
// K chunks [kb..kb+7] and [kb+16..kb+23], kb = (lane>>4)*8.
// caller passes koff = k0 + (lane>>4)*8
__device__ __forceinline__ v16h load_afrag(const h16* rowp, int koff) {
  AFrag f;
  f.h2[0] = *(const v8h*)(rowp + koff);
  f.h2[1] = *(const v8h*)(rowp + koff + 16);
  return f.v;
}

// B-fragment loader from 16-byte-aligned (not necessarily 32B) memory: K chunks [p..p+7] and [p+16..p+23].
__device__ __forceinline__ v16h load_b16(const h16* p) {
  AFrag f;
  f.h2[0] = *(const v8h*)p;
  f.h2[1] = *(const v8h*)(p + 16);
  return f.v;
}

#if PERF_ASYNC_LDS
__device__ __forceinline__ AS1 v4i* gcast(const void* p) {
  return (AS1 v4i*)(unsigned long long)p;    // global aperture is identity
}
__device__ __forceinline__ AS3 v4i* lcast(void* p) {
  return (AS3 v4i*)p;                        // addrspacecast generic->LDS
}
#endif

// ---------------------------------------------------------------- cvt f32->f16
__global__ __launch_bounds__(256) void perf_cvt(const float* __restrict__ s,
                                                h16* __restrict__ d, int n) {
  int i = blockIdx.x * 256 + threadIdx.x;
  if (i < n) d[i] = (h16)s[i];
}

// ---------------------------------------------------------------- big GEMM
// C[i,j] = sum_k A[i,k]*W[j,k] + bias[j]   (2048 x 1024 x 1024)
// mode 0: outF as f32 [B,H,T,64] (q,k)   mode 1: outH as f16 [B,H,64,T] (vT)
// mode 2: outF as f32 [i,j] flat (final projection)
// wave tile 32x64 -> 8 wmma per 12 b128 loads per k-step
__global__ __launch_bounds__(256) void perf_wgemm(
    const h16* __restrict__ A, const h16* __restrict__ W,
    const float* __restrict__ bias,
    float* __restrict__ outF, h16* __restrict__ outH, int mode)
{
  const int tid  = threadIdx.x;
  const int lane = tid & 31, wid = tid >> 5;
  const int wm = wid & 3, wn = wid >> 2;        // 4x2 wave grid
  const int l15 = lane & 15, lh = lane >> 4;
  const int akb = lh * 8, bkb = lh * 8;
  const int rowBase = blockIdx.x * 128 + wm * 32;
  const int colBase = blockIdx.y * 128 + wn * 64;

  const h16* a0p = A + (size_t)(rowBase + l15) * PC;
  const h16* a1p = A + (size_t)(rowBase + 16 + l15) * PC;
  const h16* b0p = W + (size_t)(colBase + l15) * PC;
  const h16* b1p = W + (size_t)(colBase + 16 + l15) * PC;
  const h16* b2p = W + (size_t)(colBase + 32 + l15) * PC;
  const h16* b3p = W + (size_t)(colBase + 48 + l15) * PC;

  v8f acc[2][4] = {};
  for (int k0 = 0; k0 < PC; k0 += 32) {
    __builtin_prefetch(a0p + k0 + 256, 0, 1);
    __builtin_prefetch(b0p + k0 + 256, 0, 1);
    v16h a0 = load_afrag(a0p, k0 + akb);
    v16h a1 = load_afrag(a1p, k0 + akb);
    v16h b0 = load_afrag(b0p, k0 + bkb);
    v16h b1 = load_afrag(b1p, k0 + bkb);
    v16h b2 = load_afrag(b2p, k0 + bkb);
    v16h b3 = load_afrag(b3p, k0 + bkb);
    acc[0][0] = WMMA_F32_F16(a0, b0, acc[0][0]);
    acc[0][1] = WMMA_F32_F16(a0, b1, acc[0][1]);
    acc[0][2] = WMMA_F32_F16(a0, b2, acc[0][2]);
    acc[0][3] = WMMA_F32_F16(a0, b3, acc[0][3]);
    acc[1][0] = WMMA_F32_F16(a1, b0, acc[1][0]);
    acc[1][1] = WMMA_F32_F16(a1, b1, acc[1][1]);
    acc[1][2] = WMMA_F32_F16(a1, b2, acc[1][2]);
    acc[1][3] = WMMA_F32_F16(a1, b3, acc[1][3]);
  }
#pragma unroll
  for (int mt = 0; mt < 2; ++mt)
#pragma unroll
    for (int nt = 0; nt < 4; ++nt) {
      const int j = colBase + nt * 16 + l15;
      const float bj = bias[j];
#pragma unroll
      for (int r = 0; r < 8; ++r) {
        const int i = rowBase + mt * 16 + r + 8 * lh;
        const float val = acc[mt][nt][r] + bj;
        if (mode == 2) {
          outF[(size_t)i * PC + j] = val;
        } else {
          const int b = i >> 10, t = i & 1023;
          const int hh = j >> 6, d = j & 63;
          if (mode == 0)
            outF[(((size_t)(b * PH + hh) * PT + t) << 6) + d] = val;
          else
            outH[(((size_t)(b * PH + hh) * PD + d) << 10) + t] = (h16)val;
        }
      }
    }
}

// ---------------------------------------------------------------- feature map
// src f32 [B,H,T,64] -> outp f16 [B,H,T,64] (and outpT f16 [B,H,64,T] for k)
__global__ __launch_bounds__(256) void perf_feat(
    const float* __restrict__ src, const float* __restrict__ proj,
    h16* __restrict__ outp, h16* __restrict__ outpT)
{
  __shared__ float pj[64][65];
  __shared__ float xs[4][64];
  __shared__ float xq[4][65];
  const int tid = threadIdx.x;
  for (int e = tid; e < 4096; e += 256) pj[e >> 6][e & 63] = proj[e];
  const int r = tid >> 6, m = tid & 63;
  const size_t row = (size_t)blockIdx.x * 4 + r;           // over B*H*T
  const float dn = 0.35355339059327373f;                   // 64^(-1/4)
  xs[r][m] = src[row * 64 + m] * dn;
  __syncthreads();
  float dot = 0.f, sq = 0.f;
  for (int d = 0; d < 64; ++d) {
    const float xv = xs[r][d];
    dot = fmaf(xv, pj[m][d], dot);
    sq  = fmaf(xv, xv, sq);
  }
  const float v = dot - 0.5f * sq;
  xq[r][m] = v;
  __syncthreads();
  float mx = -3.0e38f;
  for (int j = 0; j < 64; ++j) mx = fmaxf(mx, xq[r][j]);
  const float o = __expf(v - mx) + 1e-6f;
  outp[row * 64 + m] = (h16)o;
  if (outpT != nullptr) {
    const size_t bh = row >> 10;
    const int t = (int)(row & 1023);
    outpT[((bh * 64 + m) << 10) + t] = (h16)o;
  }
}

// ---------------------------------------------------------------- pass A
// per (b,h,chunk): Ssum[m,d] = sum_t k'[t,m] v[t,d]; zsum[m] = sum_t k'[t,m]
__global__ __launch_bounds__(256) void perf_chunk_sum(
    const h16* __restrict__ kpT,  // [B,H,64m,1024t]
    const h16* __restrict__ vT,   // [B,H,64d,1024t]
    float* __restrict__ Ssum,     // [B,H,16,64m*64d]
    float* __restrict__ zsum)     // [B,H,16,64]
{
  __shared__ float zpart[64][5];
  const int bh = blockIdx.x >> 4, c = blockIdx.x & 15;
  const h16* kp = kpT + (size_t)bh * PD * PT + c * 64;
  const h16* vp = vT  + (size_t)bh * PD * PT + c * 64;
  const int tid = threadIdx.x;
  const int lane = tid & 31, wid = tid >> 5;
  const int wm = wid & 3, wn = wid >> 2;
  const int l15 = lane & 15, lh = lane >> 4;
  const int akb = lh * 8, bkb = lh * 8;

  const h16* ap = kp + (size_t)(wm * 16 + l15) * PT;
  v8f acc[2] = {};
  for (int k0 = 0; k0 < 64; k0 += 32) {
    v16h a = load_afrag(ap, k0 + akb);
#pragma unroll
    for (int nt = 0; nt < 2; ++nt) {
      const int dg = (wn * 2 + nt) * 16 + l15;
      v16h b = load_afrag(vp + (size_t)dg * PT, k0 + bkb);
      acc[nt] = WMMA_F32_F16(a, b, acc[nt]);
    }
  }
  float* So = Ssum + ((size_t)bh * PNC + c) * 4096;
#pragma unroll
  for (int nt = 0; nt < 2; ++nt)
#pragma unroll
    for (int r = 0; r < 8; ++r) {
      const int m = wm * 16 + r + 8 * lh;
      const int d = (wn * 2 + nt) * 16 + l15;
      So[m * 64 + d] = acc[nt][r];
    }
  const int m2 = tid & 63, part = tid >> 6;
  float zs = 0.f;
  for (int j = 0; j < 16; ++j)
    zs += (float)kp[(size_t)m2 * PT + part * 16 + j];
  zpart[m2][part] = zs;
  __syncthreads();
  if (tid < 64)
    zsum[((size_t)bh * PNC + c) * 64 + tid] =
        zpart[tid][0] + zpart[tid][1] + zpart[tid][2] + zpart[tid][3];
}

// ---------------------------------------------------------------- pass B: scan
// exclusive prefix over the 16 chunk states; SpT stored transposed [d][m] f16
__global__ __launch_bounds__(256) void perf_scan(
    const float* __restrict__ Ssum, const float* __restrict__ zsum,
    h16* __restrict__ SpT, float* __restrict__ zprev)
{
  const int bh = blockIdx.x;
  const float* S = Ssum + (size_t)bh * PNC * 4096;
  const float* z = zsum + (size_t)bh * PNC * 64;
  h16* Sp = SpT + (size_t)bh * PNC * 4096;
  float* zp = zprev + (size_t)bh * PNC * 64;
  const int tid = threadIdx.x;
  float acc[16];
  for (int j = 0; j < 16; ++j) acc[j] = 0.f;
  float za = 0.f;
  for (int c = 0; c < PNC; ++c) {
    for (int j = 0; j < 16; ++j) {
      const int e = tid * 16 + j;          // e = m*64 + d
      const int m = e >> 6, d = e & 63;
      Sp[c * 4096 + d * 64 + m] = (h16)acc[j];
    }
    for (int j = 0; j < 16; ++j)
      acc[j] += S[c * 4096 + tid * 16 + j];
    if (tid < 64) {
      zp[c * 64 + tid] = za;
      za += z[c * 64 + tid];
    }
  }
}

// ---------------------------------------------------------------- pass C
// num = Q'_c @ Sprev + tril(Q'_c K'_c^T) @ V_c ; den = Q'.zprev + rowsum(tril A)
// K', V^T, Sprev tiles (8KB each) staged once into LDS via async-to-LDS DMA.
__global__ __launch_bounds__(256) void perf_chunk_out(
    const h16* __restrict__ qp,   // [B,H,T,64]
    const h16* __restrict__ kp,   // [B,H,T,64]
    const h16* __restrict__ vT,   // [B,H,64,T]
    const h16* __restrict__ SpT,  // [B,H,16,64d*64m]
    const float* __restrict__ zprev, // [B,H,16,64]
    h16* __restrict__ y)          // [B,T,1024], col = h*64+d
{
  __shared__ __align__(16) h16 As[64][80];
  __shared__ __align__(16) h16 kpl[64][72];   // K'_c   [s][m]
  __shared__ __align__(16) h16 vtl[64][72];   // V_c^T  [d][t]
  __shared__ __align__(16) h16 spl[64][72];   // Sprev^T[d][m]
  __shared__ float denp[64][5];
  __shared__ float den[64];
  const int bh = blockIdx.x >> 4, c = blockIdx.x & 15;
  const int b = bh >> 4, hh = bh & 15;
  const h16* qpc = qp + (size_t)bh * PT * 64 + (size_t)c * 64 * 64;
  const h16* kpc = kp + (size_t)bh * PT * 64 + (size_t)c * 64 * 64;
  const h16* vTc = vT + (size_t)bh * PD * PT + c * 64;
  const h16* Sp  = SpT + ((size_t)bh * PNC + c) * 4096;
  const float* zpv = zprev + ((size_t)bh * PNC + c) * 64;

  const int tid = threadIdx.x;
  const int lane = tid & 31, wid = tid >> 5;
  const int wm = wid & 3, wn = wid >> 2;
  const int l15 = lane & 15, lh = lane >> 4;
  const int akb = lh * 8, bkb = lh * 8;
  const h16* aqp = qpc + (size_t)(wm * 16 + l15) * 64;

  // 0) stage the three shared tiles into LDS (each thread moves 2x16B per tile)
#if PERF_ASYNC_LDS
#pragma unroll
  for (int it = 0; it < 2; ++it) {
    const int cc = it * 256 + tid;            // 0..511 chunks of 8 halves
    const int rr = cc >> 3, c8 = (cc & 7) * 8;
    __builtin_amdgcn_global_load_async_to_lds_b128(
        gcast(kpc + (size_t)rr * 64 + c8),  lcast(&kpl[rr][c8]), 0, 0);
    __builtin_amdgcn_global_load_async_to_lds_b128(
        gcast(vTc + (size_t)rr * PT + c8),  lcast(&vtl[rr][c8]), 0, 0);
    __builtin_amdgcn_global_load_async_to_lds_b128(
        gcast(Sp  + (size_t)rr * 64 + c8),  lcast(&spl[rr][c8]), 0, 0);
  }
  __builtin_amdgcn_s_wait_asynccnt(0);
  __syncthreads();
#else
#pragma unroll
  for (int it = 0; it < 2; ++it) {
    const int cc = it * 256 + tid;
    const int rr = cc >> 3, c8 = (cc & 7) * 8;
    *(v8h*)&kpl[rr][c8] = *(const v8h*)(kpc + (size_t)rr * 64 + c8);
    *(v8h*)&vtl[rr][c8] = *(const v8h*)(vTc + (size_t)rr * PT + c8);
    *(v8h*)&spl[rr][c8] = *(const v8h*)(Sp  + (size_t)rr * 64 + c8);
  }
  __syncthreads();
#endif

  // 1) A = Q'_c @ K'_c^T (K over m=64), mask s<=t, stash as f16 in LDS
  v8f aacc[2] = {};
  for (int k0 = 0; k0 < 64; k0 += 32) {
    v16h a = load_afrag(aqp, k0 + akb);
#pragma unroll
    for (int nt = 0; nt < 2; ++nt) {
      const int sg = (wn * 2 + nt) * 16 + l15;
      v16h bb = load_b16(&kpl[sg][k0 + bkb]);
      aacc[nt] = WMMA_F32_F16(a, bb, aacc[nt]);
    }
  }
#pragma unroll
  for (int nt = 0; nt < 2; ++nt)
#pragma unroll
    for (int r = 0; r < 8; ++r) {
      const int t = wm * 16 + r + 8 * lh;
      const int s = (wn * 2 + nt) * 16 + l15;
      As[t][s] = (h16)((s <= t) ? aacc[nt][r] : 0.f);
    }
  __syncthreads();

  // 2) den[t] = rowsum(As[t]) + q'[t].zprev + 1e-6
  {
    const int rr = tid >> 2, jj = tid & 3;
    float p = 0.f;
    for (int s = jj * 16; s < jj * 16 + 16; ++s) p += (float)As[rr][s];
    for (int m = jj * 16; m < jj * 16 + 16; ++m)
      p = fmaf((float)qpc[(size_t)rr * 64 + m], zpv[m], p);
    denp[rr][jj] = p;
  }
  __syncthreads();
  if (tid < 64)
    den[tid] = denp[tid][0] + denp[tid][1] + denp[tid][2] + denp[tid][3] + 1e-6f;
  __syncthreads();

  // 3) num = Q'_c @ Sprev  +  As @ V_c
  v8f nacc[2] = {};
  for (int k0 = 0; k0 < 64; k0 += 32) {
    v16h a = load_afrag(aqp, k0 + akb);
#pragma unroll
    for (int nt = 0; nt < 2; ++nt) {
      const int dg = (wn * 2 + nt) * 16 + l15;
      v16h bb = load_b16(&spl[dg][k0 + bkb]);
      nacc[nt] = WMMA_F32_F16(a, bb, nacc[nt]);
    }
  }
  const h16* arow = &As[wm * 16 + l15][0];
  for (int k0 = 0; k0 < 64; k0 += 32) {
    v16h a = load_afrag(arow, k0 + akb);          // ds_load from LDS
#pragma unroll
    for (int nt = 0; nt < 2; ++nt) {
      const int dg = (wn * 2 + nt) * 16 + l15;
      v16h bb = load_b16(&vtl[dg][k0 + bkb]);
      nacc[nt] = WMMA_F32_F16(a, bb, nacc[nt]);
    }
  }
  // 4) y = num / den
#pragma unroll
  for (int nt = 0; nt < 2; ++nt)
#pragma unroll
    for (int r = 0; r < 8; ++r) {
      const int t = wm * 16 + r + 8 * lh;
      const int d = (wn * 2 + nt) * 16 + l15;
      const int tg = c * 64 + t;
      const float val = nacc[nt][r] / den[t];
      y[((size_t)b * PT + tg) * PC + hh * PD + d] = (h16)val;
    }
}

// ---------------------------------------------------------------- launcher
extern "C" void kernel_launch(void* const* d_in, const int* in_sizes, int n_in,
                              void* d_out, int out_size, void* d_ws, size_t ws_size,
                              hipStream_t stream)
{
  const float* x    = (const float*)d_in[0];
  const float* wq   = (const float*)d_in[1];
  const float* bq   = (const float*)d_in[2];
  const float* wk   = (const float*)d_in[3];
  const float* bk   = (const float*)d_in[4];
  const float* wv   = (const float*)d_in[5];
  const float* bv   = (const float*)d_in[6];
  const float* wo   = (const float*)d_in[7];
  const float* bo   = (const float*)d_in[8];
  const float* proj = (const float*)d_in[9];

  const size_t NT = (size_t)PB * PT;        // 2048
  size_t off = 0;
  char* ws = (char*)d_ws;
  h16*   xh   = (h16*)(ws + off);  off += NT * PC * 2;          // 4 MB
  h16*   wqh  = (h16*)(ws + off);  off += (size_t)PC * PC * 2;  // 2 MB
  h16*   wkh  = (h16*)(ws + off);  off += (size_t)PC * PC * 2;
  h16*   wvh  = (h16*)(ws + off);  off += (size_t)PC * PC * 2;
  h16*   woh  = (h16*)(ws + off);  off += (size_t)PC * PC * 2;
  float* qf   = (float*)(ws + off); off += NT * PC * 4;         // 8 MB
  float* kf   = (float*)(ws + off); off += NT * PC * 4;
  h16*   vTh  = (h16*)(ws + off);  off += NT * PC * 2;
  h16*   qph  = (h16*)(ws + off);  off += NT * PC * 2;
  h16*   kph  = (h16*)(ws + off);  off += NT * PC * 2;
  h16*   kpTh = (h16*)(ws + off);  off += NT * PC * 2;
  float* Ssum = (float*)(ws + off); off += (size_t)PB * PH * PNC * 4096 * 4;
  float* zsum = (float*)(ws + off); off += (size_t)PB * PH * PNC * 64 * 4;
  h16*   SpTh = (h16*)(ws + off);  off += (size_t)PB * PH * PNC * 4096 * 2;
  float* zprv = (float*)(ws + off); off += (size_t)PB * PH * PNC * 64 * 4;
  h16*   yh   = (h16*)(ws + off);  off += NT * PC * 2;
  if (ws_size < off) return;

  const int nxy = (int)(NT * PC);           // 2M
  const int nw  = PC * PC;                  // 1M
  perf_cvt<<<(nxy + 255) / 256, 256, 0, stream>>>(x,  xh,  nxy);
  perf_cvt<<<(nw  + 255) / 256, 256, 0, stream>>>(wq, wqh, nw);
  perf_cvt<<<(nw  + 255) / 256, 256, 0, stream>>>(wk, wkh, nw);
  perf_cvt<<<(nw  + 255) / 256, 256, 0, stream>>>(wv, wvh, nw);
  perf_cvt<<<(nw  + 255) / 256, 256, 0, stream>>>(wo, woh, nw);

  dim3 gg(16, 8);                            // 128x128 block tiles
  perf_wgemm<<<gg, 256, 0, stream>>>(xh, wqh, bq, qf, nullptr, 0);
  perf_wgemm<<<gg, 256, 0, stream>>>(xh, wkh, bk, kf, nullptr, 0);
  perf_wgemm<<<gg, 256, 0, stream>>>(xh, wvh, bv, nullptr, vTh, 1);

  perf_feat<<<(PB * PH * PT) / 4, 256, 0, stream>>>(qf, proj, qph, nullptr);
  perf_feat<<<(PB * PH * PT) / 4, 256, 0, stream>>>(kf, proj, kph, kpTh);

  perf_chunk_sum<<<PB * PH * PNC, 256, 0, stream>>>(kpTh, vTh, Ssum, zsum);
  perf_scan<<<PB * PH, 256, 0, stream>>>(Ssum, zsum, SpTh, zprv);
  perf_chunk_out<<<PB * PH * PNC, 256, 0, stream>>>(qph, kph, vTh, SpTh, zprv, yh);

  perf_wgemm<<<gg, 256, 0, stream>>>(yh, woh, bo, (float*)d_out, nullptr, 2);
}
